// ScoreNet_19799799234576
// MI455X (gfx1250) — matrix operand order repaired and checked by reading, not compiled, not run on hardware
//
#include <hip/hip_runtime.h>
#include <math.h>

// ---------------- types ----------------
typedef __attribute__((ext_vector_type(16))) __bf16 v16bf;
typedef __attribute__((ext_vector_type(8)))  __bf16 v8bf;
typedef __attribute__((ext_vector_type(8)))  float  v8f;
typedef __attribute__((ext_vector_type(4)))  int    v4i;

// ---------------- CDNA5 async LDS copy support ----------------
#if __has_builtin(__builtin_amdgcn_global_load_async_to_lds_b128)
#define USE_ASYNC_LDS 1
#else
#define USE_ASYNC_LDS 0
#endif

#if __has_builtin(__builtin_amdgcn_s_wait_asynccnt)
#define WAIT_ASYNC() __builtin_amdgcn_s_wait_asynccnt(0)
#else
#define WAIT_ASYNC() asm volatile("s_wait_asynccnt 0x0" ::: "memory")
#endif

__device__ __forceinline__ __attribute__((address_space(1))) v4i* gbl_v4i(const void* p) {
  return (__attribute__((address_space(1))) v4i*)(unsigned long long)p;
}
__device__ __forceinline__ __attribute__((address_space(3))) v4i* lds_v4i(void* p) {
  return (__attribute__((address_space(3))) v4i*)(unsigned)(unsigned long long)p;
}

// ---------------- problem constants ----------------
constexpr int kB  = 1024;
constexpr int kT  = 81;
constexpr int kTP = 96;          // T padded to multiple of 16 for WMMA tiles
constexpr int kD  = 128;
constexpr int kH  = 8;
constexpr int kHD = 16;
constexpr int kE  = 256;
constexpr int kL  = 20;
constexpr int kRF = 8;
constexpr int kBT = kB * kT;     // 82944 = 64 * 1296 (exact)
constexpr size_t kQKVstride = (size_t)kB * kH * kTP * kHD;  // elems per padded q/k/v tensor
constexpr size_t kRelLayer  = (size_t)kH * kTP * kTP;       // padded rel-bias elems per layer

__device__ __forceinline__ float geluf(float x) {
  return 0.5f * x * (1.0f + tanhf(0.7978845608028654f * (x + 0.044715f * x * x * x)));
}
__device__ __forceinline__ __bf16 f2bf(float x) { return (__bf16)x; }

// ---------------- tiny elementwise kernels ----------------
__global__ void cvt_f32_bf16_kernel(const float* __restrict__ s, __bf16* __restrict__ d, int n) {
  int i = blockIdx.x * blockDim.x + threadIdx.x;
  if (i < n) d[i] = f2bf(s[i]);
}

__global__ void zero_bf16_kernel(__bf16* __restrict__ p, int n) {
  int i = blockIdx.x * blockDim.x + threadIdx.x;
  if (i < n) p[i] = f2bf(0.0f);
}

// emb_raw[b, j] = sin/cos(t[b] * W[j%128] * 2pi), bf16
__global__ void fourier_kernel(const float* __restrict__ t, const float* __restrict__ W,
                               __bf16* __restrict__ out) {
  int i = blockIdx.x * blockDim.x + threadIdx.x;
  if (i >= kB * kE) return;
  int b = i / kE, j = i - b * kE;
  float ang = t[b] * W[j < kE / 2 ? j : j - kE / 2] * 6.283185307179586f;
  out[i] = f2bf(j < kE / 2 ? sinf(ang) : cosf(ang));
}

// Padded + pre-masked relative-position bias: rel[l,h,q,k] over (kTP x kTP).
//   k >= T        -> -1e30 (softmax mask for padded keys)
//   q >= T, k < T -> 0     (padded query rows, values unused)
//   else          -> sum_f allenc[q,k,f] * bw[l,f,h] + bb[l,h]
__global__ void rel_bias_kernel(const float* __restrict__ allenc, const float* __restrict__ bw,
                                const float* __restrict__ bb, float* __restrict__ rel) {
  int i = blockIdx.x * blockDim.x + threadIdx.x;
  if (i >= kL * (int)kRelLayer) return;
  int kk = i % kTP;
  int q  = (i / kTP) % kTP;
  int h  = (i / (kTP * kTP)) % kH;
  int l  = i / (kTP * kTP * kH);
  float v;
  if (kk >= kT) {
    v = -1e30f;
  } else if (q >= kT) {
    v = 0.0f;
  } else {
    v = bb[l * kH + h];
    const float* a = allenc + (size_t)(q * kT + kk) * kRF;
    const float* w = bw + (size_t)l * kRF * kH + h;
#pragma unroll
    for (int f = 0; f < kRF; ++f) v += a[f] * w[(size_t)f * kH];
  }
  rel[i] = v;
}

// h[b,t,:] = x[b,t,:9] @ lin_w(9,128) + lin_b  (K=9 -> VALU)
__global__ void in_proj_kernel(const float* __restrict__ x, const float* __restrict__ w,
                               const float* __restrict__ bias, float* __restrict__ h) {
  int i = blockIdx.x * blockDim.x + threadIdx.x;
  if (i >= kBT * kD) return;
  int d = i & (kD - 1);
  int row = i >> 7;
  float s = bias[d];
  const float* xr = x + (size_t)row * 9;
#pragma unroll
  for (int f = 0; f < 9; ++f) s += xr[f] * w[f * kD + d];
  h[i] = s;
}

// out[bt,:9] = h[bt,:] @ out_w + out_b, minus row mean
__global__ void out_proj_kernel(const float* __restrict__ h, const float* __restrict__ ow,
                                const float* __restrict__ ob, float* __restrict__ out) {
  int row = blockIdx.x * blockDim.x + threadIdx.x;
  if (row >= kBT) return;
  float acc[9];
#pragma unroll
  for (int j = 0; j < 9; ++j) acc[j] = ob[j];
  const float* hr = h + (size_t)row * kD;
  for (int k = 0; k < kD; ++k) {
    float v = hr[k];
#pragma unroll
    for (int j = 0; j < 9; ++j) acc[j] += v * ow[k * 9 + j];
  }
  float m = 0.f;
#pragma unroll
  for (int j = 0; j < 9; ++j) m += acc[j];
  m *= (1.0f / 9.0f);
#pragma unroll
  for (int j = 0; j < 9; ++j) out[(size_t)row * 9 + j] = acc[j] - m;
}

// optional broadcast-add of c[b,:] into h, then LayerNorm over D=128 -> bf16 out.
// one wave32 per row, 4 elems per lane, wave32 shuffle reductions.
__launch_bounds__(256)
__global__ void addc_ln_kernel(float* __restrict__ h, const float* __restrict__ c,
                               const float* __restrict__ g, const float* __restrict__ beta,
                               __bf16* __restrict__ out, int withC) {
  int row  = blockIdx.x * 8 + (threadIdx.x >> 5);
  int lane = threadIdx.x & 31;
  float* hr = h + (size_t)row * kD + lane * 4;
  float4 v = *(float4*)hr;
  if (withC) {
    int b = row / kT;
    const float4 cc = *(const float4*)(c + (size_t)b * kD + lane * 4);
    v.x += cc.x; v.y += cc.y; v.z += cc.z; v.w += cc.w;
    *(float4*)hr = v;
  }
  float s = v.x + v.y + v.z + v.w;
#pragma unroll
  for (int off = 16; off; off >>= 1) s += __shfl_xor(s, off, 32);
  float mean = s * (1.0f / 128.0f);
  float dx = v.x - mean, dy = v.y - mean, dz = v.z - mean, dw = v.w - mean;
  float q = dx * dx + dy * dy + dz * dz + dw * dw;
#pragma unroll
  for (int off = 16; off; off >>= 1) q += __shfl_xor(q, off, 32);
  float inv = rsqrtf(q * (1.0f / 128.0f) + 1e-5f);
  int i = lane * 4;
  __bf16* o = out + (size_t)row * kD + i;
  o[0] = f2bf(dx * inv * g[i + 0] + beta[i + 0]);
  o[1] = f2bf(dy * inv * g[i + 1] + beta[i + 1]);
  o[2] = f2bf(dz * inv * g[i + 2] + beta[i + 2]);
  o[3] = f2bf(dw * inv * g[i + 3] + beta[i + 3]);
}

// ---------------- tiled bf16 WMMA GEMM: C(M,N) = A(M,K) @ W(K,N) + bias ----------------
// 64x64 block tile, K-step 32, 128 threads (4 waves), each wave owns a 16-row strip x 64 cols.
// A tile staged via GLOBAL_LOAD_ASYNC_TO_LDS_B128 (ASYNCcnt) when available.
enum : int { EPI_BF16 = 0, EPI_BF16_GELU = 1, EPI_F32 = 2, EPI_RESID = 3, EPI_QKV = 4 };

template <int EPI>
__launch_bounds__(128)
__global__ void gemm_bf16_kernel(const __bf16* __restrict__ A, const __bf16* __restrict__ W,
                                 const float* __restrict__ bias, float* __restrict__ outF,
                                 __bf16* __restrict__ outB, int M, int N, int K) {
  __shared__ __bf16 As[64][32];   // A tile, row-major (K contiguous)
  __shared__ __bf16 Wt[64][32];   // W tile TRANSPOSED: Wt[n][k] -> B-fragment = 1 contiguous 32B read

  const int tid  = threadIdx.x;
  const int wave = tid >> 5;
  const int lane = tid & 31;
  const int fm   = lane & 15;     // fragment row/col within 16
  const int hi   = lane >> 4;     // half-wave select
  const int m0   = blockIdx.y * 64;
  const int n0   = blockIdx.x * 64;

  v8f acc[4] = {};                // 4 n-tiles of the wave's 16-row strip

  for (int kk = 0; kk < K; kk += 32) {
    {   // stage A: 64x32 bf16, 32 bytes per thread
      int m = tid >> 1;
      int k = (tid & 1) * 16;
      const __bf16* src = A + (size_t)(m0 + m) * K + kk + k;
#if USE_ASYNC_LDS
      __builtin_amdgcn_global_load_async_to_lds_b128(gbl_v4i(src), lds_v4i(&As[m][k]), 0, 0);
      __builtin_amdgcn_global_load_async_to_lds_b128(gbl_v4i(src + 8), lds_v4i(&As[m][k + 8]), 0, 0);
#else
      *(v8bf*)&As[m][k]     = *(const v8bf*)src;
      *(v8bf*)&As[m][k + 8] = *(const v8bf*)(src + 8);
#endif
    }
    {   // stage W transposed: coalesced global read (n contiguous), scattered LDS write
      for (int c = tid; c < 256; c += 128) {
        int k  = c >> 3;
        int nb = (c & 7) * 8;
        v8bf w = *(const v8bf*)(W + (size_t)(kk + k) * N + n0 + nb);
#pragma unroll
        for (int j = 0; j < 8; ++j) Wt[nb + j][k] = w[j];
      }
    }
#if USE_ASYNC_LDS
    WAIT_ASYNC();
#endif
    __syncthreads();

    // A fragment (lane m = L&15, K = (L>>4)*16 + e), matching B fragment order
    v16bf af;
    {
      int kb = hi * 16;
      v8bf lo  = *(const v8bf*)&As[wave * 16 + fm][kb];
      v8bf hi8 = *(const v8bf*)&As[wave * 16 + fm][kb + 8];
#pragma unroll
      for (int j = 0; j < 8; ++j) { af[j] = lo[j]; af[8 + j] = hi8[j]; }
    }
#pragma unroll
    for (int nt = 0; nt < 4; ++nt) {
      // B fragment (lane n = L&15, K = (L>>4)*16 + e): one 32B LDS read from transposed tile
      v16bf bfr = *(const v16bf*)&Wt[nt * 16 + fm][hi * 16];
      acc[nt] = __builtin_amdgcn_wmma_f32_16x16x32_bf16(false, af, false, bfr,
                                                        (short)0, acc[nt], false, false);
    }
    __syncthreads();
  }

  // epilogue: C layout lane n = L&15, VGPR r -> m = r + 8*(L>>4)
#pragma unroll
  for (int nt = 0; nt < 4; ++nt) {
#pragma unroll
    for (int r = 0; r < 8; ++r) {
      int row = m0 + wave * 16 + r + 8 * hi;
      int col = n0 + nt * 16 + fm;
      float v = acc[nt][r] + bias[col];
      if (EPI == EPI_BF16) {
        outB[(size_t)row * N + col] = f2bf(v);
      } else if (EPI == EPI_BF16_GELU) {
        outB[(size_t)row * N + col] = f2bf(geluf(v));
      } else if (EPI == EPI_F32) {
        outF[(size_t)row * N + col] = v;
      } else if (EPI == EPI_RESID) {
        outF[(size_t)row * N + col] += v;
      } else { // EPI_QKV: scatter into padded (B,H,96,16) q/k/v tensors
        int which = col / kD;          // 0=q 1=k 2=v
        int d     = col - which * kD;
        int head  = d >> 4, hd = d & 15;
        int b     = row / kT, t = row - b * kT;
        __bf16* dst = outB + (size_t)which * kQKVstride;
        dst[(((size_t)(b * kH + head)) * kTP + t) * kHD + hd] = f2bf(v);
      }
    }
  }
}

// ---------------- fused attention: one block per (b,h), 3 waves, all-LDS ----------------
// S = q k^T (raw accum in LDS); softmax pass fuses *0.25 + pre-masked rel bias; Y = P V.
__launch_bounds__(96)
__global__ void attention_kernel(const __bf16* __restrict__ qbuf, const __bf16* __restrict__ kbuf,
                                 const __bf16* __restrict__ vbuf, const float* __restrict__ rel,
                                 __bf16* __restrict__ ybuf) {
  extern __shared__ char smem[];
  __bf16 (*sQ)[32]   = (__bf16(*)[32])(smem);                 //  6144 B
  __bf16 (*sK)[32]   = (__bf16(*)[32])(smem + 6144);          //  6144 B
  __bf16 (*sVt)[kTP] = (__bf16(*)[kTP])(smem + 12288);        //  3072 B (transposed v: sVt[hd][t])
  float  (*sS)[kTP]  = (float (*)[kTP])(smem + 15360);        // 36864 B
  __bf16 (*sP)[kTP]  = (__bf16(*)[kTP])(smem + 52224);        // 18432 B  -> total 70656 B

  const int bh   = blockIdx.x;          // b*H + h
  const int h    = bh & (kH - 1);
  const int b    = bh >> 3;
  const int tid  = threadIdx.x;         // 0..95
  const int wave = tid >> 5;
  const int lane = tid & 31;
  const int fm   = lane & 15;
  const int hi   = lane >> 4;

  {   // load q,k (zero-pad K 16->32), v transposed
    const __bf16* qr = qbuf + (size_t)bh * kTP * kHD + tid * kHD;
    const __bf16* kr = kbuf + (size_t)bh * kTP * kHD + tid * kHD;
    const __bf16* vr = vbuf + (size_t)bh * kTP * kHD + tid * kHD;
    v8bf z = {};
    *(v8bf*)&sQ[tid][0]  = *(const v8bf*)qr;
    *(v8bf*)&sQ[tid][8]  = *(const v8bf*)(qr + 8);
    *(v8bf*)&sQ[tid][16] = z; *(v8bf*)&sQ[tid][24] = z;
    *(v8bf*)&sK[tid][0]  = *(const v8bf*)kr;
    *(v8bf*)&sK[tid][8]  = *(const v8bf*)(kr + 8);
    *(v8bf*)&sK[tid][16] = z; *(v8bf*)&sK[tid][24] = z;
#pragma unroll
    for (int j = 0; j < kHD; ++j) sVt[j][tid] = vr[j];
  }
  __syncthreads();

  // scores: each wave handles 2 of 6 M-tiles, all 6 N-tiles; raw accum -> LDS (branch-free)
#pragma unroll
  for (int mi = 0; mi < 2; ++mi) {
    int mt = wave * 2 + mi;
    v16bf af;
    {
      int kb = hi * 16;
      v8bf lo  = *(const v8bf*)&sQ[mt * 16 + fm][kb];
      v8bf hi8 = *(const v8bf*)&sQ[mt * 16 + fm][kb + 8];
#pragma unroll
      for (int j = 0; j < 8; ++j) { af[j] = lo[j]; af[8 + j] = hi8[j]; }
    }
#pragma unroll
    for (int nt = 0; nt < 6; ++nt) {
      v16bf bfr = *(const v16bf*)&sK[nt * 16 + fm][hi * 16];  // k stored row-major == transposed k^T
      v8f acc = {};
      acc = __builtin_amdgcn_wmma_f32_16x16x32_bf16(false, af, false, bfr,
                                                    (short)0, acc, false, false);
#pragma unroll
      for (int r = 0; r < 8; ++r) {
        sS[mt * 16 + r + 8 * hi][nt * 16 + fm] = acc[r];
      }
    }
  }
  __syncthreads();

  {   // softmax over row tid; fuse scale (1/sqrt(16)) + pre-masked rel bias, float4-vectorized
    const float4* rr = (const float4*)(rel + ((size_t)h * kTP + tid) * kTP);
    float4* srow = (float4*)&sS[tid][0];
    float mx = -3.0e38f;
#pragma unroll 4
    for (int j4 = 0; j4 < kTP / 4; ++j4) {
      float4 s = srow[j4];
      float4 r4 = rr[j4];
      s.x = s.x * 0.25f + r4.x;
      s.y = s.y * 0.25f + r4.y;
      s.z = s.z * 0.25f + r4.z;
      s.w = s.w * 0.25f + r4.w;
      srow[j4] = s;
      mx = fmaxf(mx, fmaxf(fmaxf(s.x, s.y), fmaxf(s.z, s.w)));
    }
    float sum = 0.f;
#pragma unroll 4
    for (int j = 0; j < kTP; ++j) { float e = __expf(sS[tid][j] - mx); sS[tid][j] = e; sum += e; }
    float inv = 1.0f / sum;
#pragma unroll 4
    for (int j = 0; j < kTP; ++j) sP[tid][j] = f2bf(sS[tid][j] * inv);
  }
  __syncthreads();

  // Y = P(96x96) @ V(96x16): K = 96 -> 3 WMMA K-chunks
#pragma unroll
  for (int mi = 0; mi < 2; ++mi) {
    int mt = wave * 2 + mi;
    v8f acc = {};
#pragma unroll
    for (int kc = 0; kc < 3; ++kc) {
      v16bf af;
      {
        int kb = kc * 32 + hi * 16;
        v8bf lo  = *(const v8bf*)&sP[mt * 16 + fm][kb];
        v8bf hi8 = *(const v8bf*)&sP[mt * 16 + fm][kb + 8];
#pragma unroll
        for (int j = 0; j < 8; ++j) { af[j] = lo[j]; af[8 + j] = hi8[j]; }
      }
      v16bf bfr = *(const v16bf*)&sVt[fm][kc * 32 + hi * 16];
      acc = __builtin_amdgcn_wmma_f32_16x16x32_bf16(false, af, false, bfr,
                                                    (short)0, acc, false, false);
    }
#pragma unroll
    for (int r = 0; r < 8; ++r) {
      int t = mt * 16 + r + 8 * hi;
      if (t < kT)
        ybuf[((size_t)b * kT + t) * kD + h * kHD + fm] = f2bf(acc[r]);
    }
  }
}

// ---------------- host-side orchestration ----------------
extern "C" void kernel_launch(void* const* d_in, const int* in_sizes, int n_in,
                              void* d_out, int out_size, void* d_ws, size_t ws_size,
                              hipStream_t stream) {
  const float* x        = (const float*)d_in[0];
  const float* t_in     = (const float*)d_in[1];
  const float* allenc   = (const float*)d_in[2];
  const float* fourierW = (const float*)d_in[3];
  const float* embed_w  = (const float*)d_in[4];
  const float* embed_b  = (const float*)d_in[5];
  const float* lin_w    = (const float*)d_in[6];
  const float* lin_b    = (const float*)d_in[7];
  const float* ln1_g    = (const float*)d_in[8];
  const float* ln1_b    = (const float*)d_in[9];
  const float* qkv_w    = (const float*)d_in[10];
  const float* qkv_b    = (const float*)d_in[11];
  const float* aproj_w  = (const float*)d_in[12];
  const float* aproj_b  = (const float*)d_in[13];
  const float* bproj_w  = (const float*)d_in[14];
  const float* bproj_b  = (const float*)d_in[15];
  const float* ln2_g    = (const float*)d_in[16];
  const float* ln2_b    = (const float*)d_in[17];
  const float* fc_w     = (const float*)d_in[18];
  const float* fc_b     = (const float*)d_in[19];
  const float* mproj_w  = (const float*)d_in[20];
  const float* mproj_b  = (const float*)d_in[21];
  const float* dense_w  = (const float*)d_in[22];
  const float* dense_b  = (const float*)d_in[23];
  const float* out_w    = (const float*)d_in[24];
  const float* out_b    = (const float*)d_in[25];
  float* out = (float*)d_out;

  char* base = (char*)d_ws;
  size_t cur = 0;
  auto takeB = [&](size_t n) -> __bf16* { __bf16* p = (__bf16*)(base + cur); cur = (cur + n * 2 + 255) & ~(size_t)255; return p; };
  auto takeF = [&](size_t n) -> float*  { float*  p = (float*)(base + cur);  cur = (cur + n * 4 + 255) & ~(size_t)255; return p; };

  // bf16 weights
  __bf16* we_bf = takeB((size_t)kE * kE);
  __bf16* wq_bf = takeB((size_t)kL * kD * 3 * kD);
  __bf16* wa_bf = takeB((size_t)kL * kD * kD);
  __bf16* wf_bf = takeB((size_t)kL * kD * 4 * kD);
  __bf16* wm_bf = takeB((size_t)kL * 4 * kD * kD);
  __bf16* wd_bf = takeB((size_t)kL * kE * kD);
  // activations / scratch
  __bf16* emb_raw = takeB((size_t)kB * kE);
  __bf16* emb_bf  = takeB((size_t)kB * kE);
  float*  c_f32   = takeF((size_t)kB * kD);
  float*  rel_all = takeF((size_t)kL * kRelLayer);
  float*  h       = takeF((size_t)kBT * kD);
  __bf16* abuf    = takeB((size_t)kBT * kD);
  __bf16* ybuf    = takeB((size_t)kBT * kD);
  __bf16* fcbuf   = takeB((size_t)kBT * 4 * kD);
  __bf16* qkvbuf  = takeB(3 * kQKVstride);    // q | k | v, padded (B,H,96,16)

  auto cvt = [&](const float* s, __bf16* d, size_t n) {
    cvt_f32_bf16_kernel<<<dim3((unsigned)((n + 255) / 256)), dim3(256), 0, stream>>>(s, d, (int)n);
  };
  cvt(embed_w, we_bf, (size_t)kE * kE);
  cvt(qkv_w,   wq_bf, (size_t)kL * kD * 3 * kD);
  cvt(aproj_w, wa_bf, (size_t)kL * kD * kD);
  cvt(fc_w,    wf_bf, (size_t)kL * kD * 4 * kD);
  cvt(mproj_w, wm_bf, (size_t)kL * 4 * kD * kD);
  cvt(dense_w, wd_bf, (size_t)kL * kE * kD);

  {   // zero padded q/k/v (padding rows must stay zero; also clears 0xAA poison)
    int n = (int)(3 * kQKVstride);
    zero_bf16_kernel<<<dim3((n + 255) / 256), dim3(256), 0, stream>>>(qkvbuf, n);
  }

  // time embedding: fourier -> GEMM(1024x256x256) + GELU
  fourier_kernel<<<dim3((kB * kE + 255) / 256), dim3(256), 0, stream>>>(t_in, fourierW, emb_raw);
  gemm_bf16_kernel<EPI_BF16_GELU><<<dim3(kE / 64, kB / 64), dim3(128), 0, stream>>>(
      emb_raw, we_bf, embed_b, nullptr, emb_bf, kB, kE, kE);

  // padded + pre-masked relative-position bias for all layers
  {
    int n = kL * (int)kRelLayer;
    rel_bias_kernel<<<dim3((n + 255) / 256), dim3(256), 0, stream>>>(allenc, bproj_w, bproj_b, rel_all);
  }

  // input projection 9 -> 128
  in_proj_kernel<<<dim3((kBT * kD + 255) / 256), dim3(256), 0, stream>>>(x, lin_w, lin_b, h);

  constexpr unsigned kAttSmem = 70656;
  for (int l = 0; l < kL; ++l) {
    // time conditioning: c = emb @ dense_w[l] + db   (1024x128, K=256)
    gemm_bf16_kernel<EPI_F32><<<dim3(kD / 64, kB / 64), dim3(128), 0, stream>>>(
        emb_bf, wd_bf + (size_t)l * kE * kD, dense_b + (size_t)l * kD, c_f32, nullptr, kB, kD, kE);
    // h += c (broadcast over T), then LN1 -> abuf(bf16)
    addc_ln_kernel<<<dim3(kBT / 8), dim3(256), 0, stream>>>(
        h, c_f32, ln1_g + (size_t)l * kD, ln1_b + (size_t)l * kD, abuf, 1);
    // qkv GEMM (82944 x 384, K=128) scattered into padded q/k/v
    gemm_bf16_kernel<EPI_QKV><<<dim3(3 * kD / 64, kBT / 64), dim3(128), 0, stream>>>(
        abuf, wq_bf + (size_t)l * kD * 3 * kD, qkv_b + (size_t)l * 3 * kD,
        nullptr, qkvbuf, kBT, 3 * kD, kD);
    // fused attention
    attention_kernel<<<dim3(kB * kH), dim3(96), kAttSmem, stream>>>(
        qkvbuf, qkvbuf + kQKVstride, qkvbuf + 2 * kQKVstride,
        rel_all + (size_t)l * kRelLayer, ybuf);
    // attention out projection, residual-add into h
    gemm_bf16_kernel<EPI_RESID><<<dim3(kD / 64, kBT / 64), dim3(128), 0, stream>>>(
        ybuf, wa_bf + (size_t)l * kD * kD, aproj_b + (size_t)l * kD, h, nullptr, kBT, kD, kD);
    // LN2 -> abuf
    addc_ln_kernel<<<dim3(kBT / 8), dim3(256), 0, stream>>>(
        h, nullptr, ln2_g + (size_t)l * kD, ln2_b + (size_t)l * kD, abuf, 0);
    // MLP: fc (128 -> 512) + GELU -> bf16
    gemm_bf16_kernel<EPI_BF16_GELU><<<dim3(4 * kD / 64, kBT / 64), dim3(128), 0, stream>>>(
        abuf, wf_bf + (size_t)l * kD * 4 * kD, fc_b + (size_t)l * 4 * kD,
        nullptr, fcbuf, kBT, 4 * kD, kD);
    // mproj (512 -> 128), residual-add into h
    gemm_bf16_kernel<EPI_RESID><<<dim3(kD / 64, kBT / 64), dim3(128), 0, stream>>>(
        fcbuf, wm_bf + (size_t)l * 4 * kD * kD, mproj_b + (size_t)l * kD, h, nullptr, kBT, kD, 4 * kD);
  }

  // output projection 128 -> 9 with mean subtraction
  out_proj_kernel<<<dim3((kBT + 255) / 256), dim3(256), 0, stream>>>(h, out_w, out_b, out);
}
